// BiLSTM_CRF_4243427688679
// MI455X (gfx1250) — matrix operand order repaired and checked by reading, not compiled, not run on hardware
//
#include <hip/hip_runtime.h>
#include <cstdint>
#include <cstddef>

#define SLEN 1024
#define BATCH 64
#define EMB 256
#define HID 128
#define G4H 512
#define NTAG 4

typedef __attribute__((ext_vector_type(16))) __bf16 v16bf;
typedef __attribute__((ext_vector_type(8)))  float  v8f;

union ABChunk { v16bf v; uint4 q[2]; };
union V8FQ    { v8f v; float4 f4[2]; };

__device__ __forceinline__ v8f zero_v8f() {
  v8f z;
#pragma unroll
  for (int i = 0; i < 8; ++i) z[i] = 0.0f;
  return z;
}

// gfx1250 has a hardware v_tanh_f32 (TRANS32). Use it when the toolchain
// exposes the builtin; sigmoid is expressed through tanh so the entire gate
// nonlinearity costs 5 TRANS32 ops + a few FMAs per element.
#if __has_builtin(__builtin_amdgcn_tanhf)
__device__ __forceinline__ float fast_tanh(float x) { return __builtin_amdgcn_tanhf(x); }
#else
__device__ __forceinline__ float fast_tanh(float x) { return tanhf(x); }
#endif
__device__ __forceinline__ float sigm(float x) { return 0.5f * fast_tanh(0.5f * x) + 0.5f; }

// ---------------------------------------------------------------------------
// Kernel 1: embedding gather, f32 -> bf16, rows laid out r = s*B + b
// ---------------------------------------------------------------------------
__global__ void k_gather_embed(const int* __restrict__ sent,
                               const float* __restrict__ embed,
                               __bf16* __restrict__ X) {
  const int r = blockIdx.x;        // r = s*B + b
  const int s = r >> 6;
  const int b = r & 63;
  const int tok = sent[b * SLEN + s];
  const float* src = embed + (size_t)tok * EMB;
  __bf16* dst = X + (size_t)r * EMB;
  dst[threadIdx.x] = (__bf16)src[threadIdx.x];
}

// ---------------------------------------------------------------------------
// Kernel 2: input projection for BOTH directions:
//   Z[dir][tile][lane][8] = (X . w_ih_dir^T + b_ih + b_hh)  stored in the
//   WMMA C register layout ("swizzled"), so the recurrent kernel can init its
//   accumulators with two b128 loads per tile.
//   tile = (p*4 + mt)*32 + ngrp, one wave's 64-row group == one time pos p.
// ---------------------------------------------------------------------------
__global__ void k_input_proj(const __bf16* __restrict__ X,
                             const float* __restrict__ w_ih_f,
                             const float* __restrict__ b_ih_f,
                             const float* __restrict__ b_hh_f,
                             const float* __restrict__ w_ih_r,
                             const float* __restrict__ b_ih_r,
                             const float* __restrict__ b_hh_r,
                             float* __restrict__ Z) {
  const int lane = threadIdx.x & 31;
  const int wid  = blockIdx.x * 8 + (threadIdx.x >> 5);  // 65536 wave-jobs
  const int dir  = wid >> 15;
  const int rjob = wid & 32767;
  const int ngrp = rjob & 31;   // N tile 0..31
  const int mgrp = rjob >> 5;   // 0..1023 == time position p
  const int rb  = mgrp * 64;
  const int col = lane & 15;
  const int hi  = lane >> 4;
  const int n   = ngrp * 16 + col;

  const float* w_ih = dir ? w_ih_r : w_ih_f;
  const float* b_ih = dir ? b_ih_r : b_ih_f;
  const float* b_hh = dir ? b_hh_r : b_hh_f;
  float* Zd = Z + (size_t)dir * ((size_t)SLEN * BATCH * G4H);

  v8f acc[4];
#pragma unroll
  for (int m = 0; m < 4; ++m) acc[m] = zero_v8f();

#pragma unroll 2
  for (int kc = 0; kc < 8; ++kc) {
    // B operand: per lane = w_ih[n][kc*32 + hi*16 .. +15] (contiguous f32 -> bf16)
    v16bf bm;
    const float* wrow = w_ih + (size_t)n * EMB + kc * 32 + hi * 16;
#pragma unroll
    for (int t = 0; t < 16; ++t) bm[t] = (__bf16)wrow[t];
#pragma unroll
    for (int m = 0; m < 4; ++m) {
      ABChunk a;
      const __bf16* arow = X + (size_t)(rb + m * 16 + col) * EMB + kc * 32 + hi * 16;
      a.q[0] = *(const uint4*)(arow);
      a.q[1] = *(const uint4*)(arow + 8);
      acc[m] = __builtin_amdgcn_wmma_f32_16x16x32_bf16(false, a.v, false, bm,
                                                       (short)0, acc[m], false, false);
    }
  }

  const float bias = b_ih[n] + b_hh[n];   // folded here; recurrent kernel adds none
#pragma unroll
  for (int m = 0; m < 4; ++m) {
    V8FQ o;
    o.v = acc[m];
#pragma unroll
    for (int j = 0; j < 8; ++j) o.v[j] += bias;
    // swizzled store: per-lane 32B contiguous, 1KB/tile fully coalesced
    float* dst = Zd + (((size_t)(mgrp * 4 + m) * 32 + ngrp) * 32 + lane) * 8;
    *(float4*)(dst)     = o.f4[0];
    *(float4*)(dst + 4) = o.f4[1];
  }
}

// ---------------------------------------------------------------------------
// Kernel 3: persistent LSTM recurrence. grid.x = 2 (one WG per direction).
// 16 waves; wave owns gate-column strip st (16 hidden cols, all 4 gates) and
// half the batch tiles -> gate mixing is wave-local in registers.
// LDS: w_hh bf16 [512][128] (128KB) + h double buffer [2][64][128] bf16 (32KB)
// Z is consumed in swizzled C-layout: 2 x b128 loads per accumulator tile.
// ---------------------------------------------------------------------------
__global__ void __launch_bounds__(512, 1)
k_lstm_recurrent(const float* __restrict__ Z,
                 const float* __restrict__ w_hh_f,
                 const float* __restrict__ w_hh_r,
                 __bf16* __restrict__ hf,
                 __bf16* __restrict__ hr) {
  extern __shared__ char smem[];
  __bf16* w_lds = (__bf16*)smem;               // [512][128] row-major (B-operand order)
  __bf16* h_lds = (__bf16*)(smem + 131072);    // [2][64][128]

  const int dir = blockIdx.x;
  const float* w_hh = dir ? w_hh_r : w_hh_f;
  __bf16* hout = dir ? hr : hf;
  const float* Zd = Z + (size_t)dir * ((size_t)SLEN * BATCH * G4H);

  const int tid = threadIdx.x;
  for (int i = tid; i < G4H * HID; i += 512) w_lds[i] = (__bf16)w_hh[i];
  for (int i = tid; i < 2 * BATCH * HID; i += 512) h_lds[i] = (__bf16)0.0f;
  __syncthreads();

  const int lane = tid & 31;
  const int wave = tid >> 5;
  const int st = wave & 7;     // hidden-column strip (16 cols)
  const int mh = wave >> 3;    // batch-tile half: tiles {2mh, 2mh+1}
  const int col = lane & 15;
  const int hi  = lane >> 4;

  int ncol[4];
#pragma unroll
  for (int g = 0; g < 4; ++g) ncol[g] = g * HID + st * 16 + col;  // i,f,g,o columns

  v8f creg[2];
  creg[0] = zero_v8f();
  creg[1] = zero_v8f();

  for (int s = 0; s < SLEN; ++s) {
    const int p = dir ? (SLEN - 1 - s) : s;     // time position consumed/produced
    // swizzled Z base for this step: 128 tiles x 256 floats
    const float* Zs = Zd + (size_t)p * (BATCH * G4H);
    const int cur = s & 1;
    const __bf16* hcur = h_lds + cur * (BATCH * HID);
    __bf16* hnxt = h_lds + (cur ^ 1) * (BATCH * HID);

    // prefetch next step's Z tiles (global_prefetch_b8; lanes cover the tiles)
    {
      const int pn = dir ? (p - 1) : (p + 1);
      if (pn >= 0 && pn < SLEN) {
        const float* Zn = Zd + (size_t)pn * (BATCH * G4H);
#pragma unroll
        for (int g = 0; g < 4; ++g)
#pragma unroll
          for (int m = 0; m < 2; ++m) {
            const int mt = mh * 2 + m;
            __builtin_prefetch(
                Zn + (((size_t)mt * 32 + (g * 8 + st)) * 32 + lane) * 8, 0, 0);
          }
      }
    }

    // accumulators init from swizzled input projection (bias already folded)
    v8f acc[4][2];
#pragma unroll
    for (int g = 0; g < 4; ++g)
#pragma unroll
      for (int m = 0; m < 2; ++m) {
        const int mt = mh * 2 + m;
        const float* src = Zs + (((size_t)mt * 32 + (g * 8 + st)) * 32 + lane) * 8;
        V8FQ a;
        a.f4[0] = *(const float4*)(src);
        a.f4[1] = *(const float4*)(src + 4);
        acc[g][m] = a.v;
      }

    // h_prev @ w_hh^T : K = 128 in 4 WMMA chunks, 8 output tiles per wave
#pragma unroll
    for (int kc = 0; kc < 4; ++kc) {
      ABChunk am[2];
#pragma unroll
      for (int m = 0; m < 2; ++m) {
        const int mt = mh * 2 + m;
        const __bf16* ap = hcur + (size_t)(mt * 16 + col) * HID + kc * 32 + hi * 16;
        am[m].q[0] = *(const uint4*)(ap);
        am[m].q[1] = *(const uint4*)(ap + 8);
      }
#pragma unroll
      for (int g = 0; g < 4; ++g) {
        ABChunk bm;
        const __bf16* bp = w_lds + (size_t)ncol[g] * HID + kc * 32 + hi * 16;
        bm.q[0] = *(const uint4*)(bp);
        bm.q[1] = *(const uint4*)(bp + 8);
#pragma unroll
        for (int m = 0; m < 2; ++m)
          acc[g][m] = __builtin_amdgcn_wmma_f32_16x16x32_bf16(
              false, am[m].v, false, bm.v, (short)0, acc[g][m], false, false);
      }
    }

    // gate math: wave-local (same (row,col) element across the 4 gate accs);
    // 5 hardware tanh + a few FMAs per element.
#pragma unroll
    for (int m = 0; m < 2; ++m) {
      const int mt = mh * 2 + m;
#pragma unroll
      for (int j = 0; j < 8; ++j) {
        const float gi = sigm(acc[0][m][j]);
        const float gf = sigm(acc[1][m][j]);
        const float gg = fast_tanh(acc[2][m][j]);
        const float go = sigm(acc[3][m][j]);
        const float c  = gf * creg[m][j] + gi * gg;
        creg[m][j] = c;
        const float h = go * fast_tanh(c);
        const __bf16 hb = (__bf16)h;
        const int row = mt * 16 + j + hi * 8;
        const int hc  = st * 16 + col;
        hnxt[row * HID + hc] = hb;
        hout[((size_t)p * BATCH + row) * HID + hc] = hb;
      }
    }
    __syncthreads();
  }
}

// ---------------------------------------------------------------------------
// Kernel 4: emissions. feats[r][t] = [hf[r] ; hr[r]] . w_out[t] + b_out[t]
// N padded to one 16-wide WMMA tile (cols 4..15 zero weights).
// ---------------------------------------------------------------------------
__global__ void k_feats(const __bf16* __restrict__ hf,
                        const __bf16* __restrict__ hr,
                        const float* __restrict__ w_out,
                        const float* __restrict__ b_out,
                        float* __restrict__ feats) {
  const int lane = threadIdx.x & 31;
  const int wv = blockIdx.x * 8 + (threadIdx.x >> 5);
  const int rb = wv * 16;
  const int col = lane & 15;
  const int hi  = lane >> 4;

  v8f acc = zero_v8f();
#pragma unroll
  for (int kc = 0; kc < 8; ++kc) {
    const __bf16* src = (kc < 4) ? hf : hr;     // concat [hf ; hr] along K
    const int kl = (kc & 3) * 32;
    v16bf bm;
    if (col < NTAG) {
      const float* wr = w_out + (size_t)col * 256 + kc * 32 + hi * 16;
#pragma unroll
      for (int t = 0; t < 16; ++t) bm[t] = (__bf16)wr[t];
    } else {
#pragma unroll
      for (int t = 0; t < 16; ++t) bm[t] = (__bf16)0.0f;
    }
    ABChunk a;
    const __bf16* ap = src + (size_t)(rb + col) * HID + kl + hi * 16;
    a.q[0] = *(const uint4*)(ap);
    a.q[1] = *(const uint4*)(ap + 8);
    acc = __builtin_amdgcn_wmma_f32_16x16x32_bf16(false, a.v, false, bm,
                                                  (short)0, acc, false, false);
  }
  if (col < NTAG) {
    const float bo = b_out[col];
#pragma unroll
    for (int j = 0; j < 8; ++j)
      feats[(size_t)(rb + j + hi * 8) * NTAG + col] = acc[j] + bo;
  }
}

// ---------------------------------------------------------------------------
// Kernel 5: Viterbi decode. One lane per batch element; bp packed 2 bits/tag.
// ---------------------------------------------------------------------------
__global__ void k_viterbi(const float* __restrict__ feats,   // [S*B][4]
                          const float* __restrict__ start_t,
                          const float* __restrict__ end_t,
                          const float* __restrict__ trans,   // [4][4]
                          unsigned char* __restrict__ bp,    // [S][B]
                          int* __restrict__ out) {           // [B][S] int32
  const int b = threadIdx.x;
  if (b >= BATCH) return;
  float tr[16];
#pragma unroll
  for (int i = 0; i < 16; ++i) tr[i] = trans[i];
  float score[4];
#pragma unroll
  for (int t = 0; t < 4; ++t) score[t] = start_t[t] + feats[(size_t)b * 4 + t];

  for (int s = 1; s < SLEN; ++s) {
    const float* fs = feats + ((size_t)s * BATCH + b) * 4;
    float ns[4];
    unsigned char pk = 0;
#pragma unroll
    for (int t = 0; t < 4; ++t) {
      float best = score[0] + tr[t];
      int bi = 0;
#pragma unroll
      for (int p = 1; p < 4; ++p) {
        const float v = score[p] + tr[p * 4 + t];
        if (v > best) { best = v; bi = p; }   // first-max wins, matches argmax
      }
      ns[t] = best + fs[t];
      pk |= (unsigned char)(bi << (2 * t));
    }
#pragma unroll
    for (int t = 0; t < 4; ++t) score[t] = ns[t];
    bp[(size_t)s * BATCH + b] = pk;
  }

  int cur = 0;
  float best = score[0] + end_t[0];
#pragma unroll
  for (int t = 1; t < 4; ++t) {
    const float v = score[t] + end_t[t];
    if (v > best) { best = v; cur = t; }
  }
  out[(size_t)b * SLEN + (SLEN - 1)] = cur;
  for (int s = SLEN - 1; s >= 1; --s) {
    const int prev = (bp[(size_t)s * BATCH + b] >> (2 * cur)) & 3;
    out[(size_t)b * SLEN + (s - 1)] = prev;
    cur = prev;
  }
}

// ---------------------------------------------------------------------------
extern "C" void kernel_launch(void* const* d_in, const int* in_sizes, int n_in,
                              void* d_out, int out_size, void* d_ws, size_t ws_size,
                              hipStream_t stream) {
  const int*   sent    = (const int*)d_in[0];
  const float* embed   = (const float*)d_in[1];
  const float* w_ih_f  = (const float*)d_in[2];
  const float* w_hh_f  = (const float*)d_in[3];
  const float* b_ih_f  = (const float*)d_in[4];
  const float* b_hh_f  = (const float*)d_in[5];
  const float* w_ih_r  = (const float*)d_in[6];
  const float* w_hh_r  = (const float*)d_in[7];
  const float* b_ih_r  = (const float*)d_in[8];
  const float* b_hh_r  = (const float*)d_in[9];
  const float* w_out   = (const float*)d_in[10];
  const float* b_out   = (const float*)d_in[11];
  const float* start_t = (const float*)d_in[12];
  const float* end_t   = (const float*)d_in[13];
  const float* trans   = (const float*)d_in[14];

  // Workspace layout (bytes)
  char* ws = (char*)d_ws;
  const size_t OFF_X   = 0;                         // X  bf16 [S*B][256]   33,554,432
  const size_t OFF_Z   = OFF_X + 33554432ull;       // Z  f32  [2][S*B*512] 268,435,456 (swizzled)
  const size_t OFF_HF  = OFF_Z + 268435456ull;      // hf bf16 [S*B][128]   16,777,216
  const size_t OFF_HR  = OFF_HF + 16777216ull;      // hr bf16 [S*B][128]   16,777,216
  const size_t OFF_FT  = OFF_HR + 16777216ull;      // feats f32 [S*B][4]    1,048,576
  const size_t OFF_BP  = OFF_FT + 1048576ull;       // bp u8 [S][B]             65,536

  __bf16* X = (__bf16*)(ws + OFF_X);
  float*  Z = (float*)(ws + OFF_Z);
  __bf16* hf = (__bf16*)(ws + OFF_HF);
  __bf16* hr = (__bf16*)(ws + OFF_HR);
  float*  feats = (float*)(ws + OFF_FT);
  unsigned char* bp = (unsigned char*)(ws + OFF_BP);

  k_gather_embed<<<SLEN * BATCH, 256, 0, stream>>>(sent, embed, X);
  k_input_proj<<<8192, 256, 0, stream>>>(X, w_ih_f, b_ih_f, b_hh_f,
                                         w_ih_r, b_ih_r, b_hh_r, Z);
  const size_t lds_bytes = 131072 + 32768;          // 160 KB < 320 KB/WGP
  hipFuncSetAttribute((const void*)k_lstm_recurrent,
                      hipFuncAttributeMaxDynamicSharedMemorySize, (int)lds_bytes);
  k_lstm_recurrent<<<2, 512, lds_bytes, stream>>>(Z, w_hh_f, w_hh_r, hf, hr);
  k_feats<<<512, 256, 0, stream>>>(hf, hr, w_out, b_out, feats);
  k_viterbi<<<1, 64, 0, stream>>>(feats, start_t, end_t, trans, bp, (int*)d_out);
}
